// HGTVulNodeClassifier_33569464386233
// MI455X (gfx1250) — matrix operand order repaired and checked by reading, not compiled, not run on hardware
//
#include <hip/hip_runtime.h>
#include <cmath>

typedef __attribute__((ext_vector_type(16))) _Float16 v16h;
typedef __attribute__((ext_vector_type(8)))  _Float16 v8h;
typedef __attribute__((ext_vector_type(8)))  float    v8f;
typedef __attribute__((ext_vector_type(4)))  unsigned v4u;
typedef __attribute__((ext_vector_type(8)))  unsigned v8u;

#define D_DIM 256
#define H_HEADS 8
#define DK_DIM 32

// ---------------------------------------------------------------------------
// Small prep kernels
// ---------------------------------------------------------------------------

// Composed relation weight, stored TRANSPOSED as f16:
// WT[r][eg][din] = sum_d W[st][din][h*32+d] * rel[r][h][d][e]   (eg = h*32+e)
__global__ void compose_rel_weight(const float* __restrict__ W,
                                   const float* __restrict__ rel,
                                   _Float16* __restrict__ WT, int n) {
  int idx = blockIdx.x * blockDim.x + threadIdx.x;
  if (idx >= n) return;
  int din = idx & 255;
  int eg  = (idx >> 8) & 255;
  int r   = idx >> 16;
  int st = r >> 1;                       // RELATIONS: st = r/2, dt = r&1
  int hh = eg >> 5, e = eg & 31;
  const float* w = W + (size_t)st * 65536 + (size_t)din * 256 + hh * 32;
  const float* m = rel + ((size_t)(r * 8 + hh) * 32) * 32 + e;
  float acc = 0.f;
#pragma unroll
  for (int d = 0; d < 32; ++d) acc += w[d] * m[(size_t)d * 32];
  WT[(size_t)(r * 256 + eg) * 256 + din] = (_Float16)acc;
}

// bout[r][eg] = sum_d b[st][h*32+d] * rel[r][h][d][e]
__global__ void compose_rel_bias(const float* __restrict__ b,
                                 const float* __restrict__ rel,
                                 float* __restrict__ bout, int n) {
  int idx = blockIdx.x * blockDim.x + threadIdx.x;
  if (idx >= n) return;
  int eg = idx & 255;
  int r  = idx >> 8;
  int st = r >> 1;
  int hh = eg >> 5, e = eg & 31;
  const float* bp = b + st * 256 + hh * 32;
  const float* m = rel + ((size_t)(r * 8 + hh) * 32) * 32 + e;
  float acc = 0.f;
#pragma unroll
  for (int d = 0; d < 32; ++d) acc += bp[d] * m[(size_t)d * 32];
  bout[idx] = acc;
}

// WT[t][nn][kk] = (f16) W[t][kk][nn]  (transpose + convert)
__global__ void transpose_w_f16(const float* __restrict__ W,
                                _Float16* __restrict__ WT, int n) {
  int idx = blockIdx.x * blockDim.x + threadIdx.x;
  if (idx >= n) return;
  int kk = idx & 255;
  int nn = (idx >> 8) & 255;
  int t  = idx >> 16;
  WT[idx] = (_Float16)W[(size_t)t * 65536 + (size_t)kk * 256 + nn];
}

__global__ void cvt2_f16(const float* __restrict__ a, const float* __restrict__ b,
                         _Float16* __restrict__ o, int nPer) {
  int i = blockIdx.x * blockDim.x + threadIdx.x;
  if (i >= 2 * nPer) return;
  float v = (i < nPer) ? a[i] : b[i - nPer];
  o[i] = (_Float16)v;
}

__global__ void scale_cvt_f16(const float* __restrict__ a, _Float16* __restrict__ o,
                              float s, int n) {
  int i = blockIdx.x * blockDim.x + threadIdx.x;
  if (i >= n) return;
  o[i] = (_Float16)(a[i] * s);
}

__global__ void fill_f32(float* __restrict__ p, float v, int n) {
  int i = blockIdx.x * blockDim.x + threadIdx.x;
  if (i < n) p[i] = v;
}

// ---------------------------------------------------------------------------
// WMMA GEMM:  C[M,256] f32 = A16[M,256] (row-major f16) * B + bias
// B given as BT16[256,256] = B^T (row n = column n of B), f16.
// Block = 128 threads = 4 waves; each wave owns a 16-row strip x 64 cols.
// The 64x256 BT tile (contiguous 32KB) is staged to LDS by the Tensor Data
// Mover: one D# descriptor, one tensor_load_to_lds, tracked on TENSORcnt.
// ---------------------------------------------------------------------------
__global__ __launch_bounds__(128) void wmma_gemm(
    const _Float16* __restrict__ A, const _Float16* __restrict__ BT,
    const float* __restrict__ bias, float* __restrict__ C, int M) {
  __shared__ __align__(32) _Float16 tileB[64 * 256];

  const int tid = threadIdx.x;
  const int n0 = blockIdx.y * 64;

  // ---- TDM: DMA the contiguous 32KB BT tile (rows n0..n0+63) into LDS ----
  if (tid < 32) {  // wave 0 issues the descriptor (EXEC is ignored by TDM)
    unsigned ldsBase = (unsigned)(size_t)(void*)tileB;
    unsigned long long gaddr =
        (unsigned long long)(const void*)(BT + (size_t)n0 * 256);
    // D# group0: count=1 | lds_addr | global_addr[56:0] | type=2
    v4u g0;
    g0.x = 1u;                                   // count=1, user mode
    g0.y = ldsBase;                              // lds_addr (bytes)
    g0.z = (unsigned)gaddr;                      // global_addr[31:0]
    g0.w = (unsigned)((gaddr >> 32) & 0x01FFFFFFull) | 0x80000000u; // [56:32]|type=2
    // D# group1: data_size=2B; tensor_dim0 = tile_dim0 = 16384 elements (1 row)
    v8u g1;
    g1.s0 = 0x00010000u;   // workgroup_mask=0, data_size=1 (2 bytes)
    g1.s1 = 0x40000000u;   // tensor_dim0[15:0]=0x4000 in bits[63:48]
    g1.s2 = 0x00010000u;   // tensor_dim0[31:16]=0; tensor_dim1[15:0]=1
    g1.s3 = 0x40000000u;   // tensor_dim1[31:16]=0; tile_dim0=0x4000
    g1.s4 = 0x00000001u;   // tile_dim1=1, tile_dim2=0
    g1.s5 = 0x00004000u;   // tensor_dim0_stride = 16384
    g1.s6 = 0u;
    g1.s7 = 0u;
    asm volatile("tensor_load_to_lds %0, %1" :: "s"(g0), "s"(g1) : "memory");
    __builtin_amdgcn_s_wait_tensorcnt(0);
  }
  __syncthreads();

  const int wid = tid >> 5;
  const int lane = tid & 31;
  const int strips = (M + 15) >> 4;
  const int strip = blockIdx.x * 4 + wid;
  if (strip >= strips) return;  // whole-wave uniform exit (after barrier)

  const int m0 = strip * 16;
  int arow = m0 + (lane & 15);
  if (arow >= M) arow = M - 1;                    // clamp (store is guarded)
  const int koff = (lane < 16) ? 0 : 16;          // A: lane half h holds k 16h..16h+15
  const _Float16* Ap = A + (size_t)arow * 256;
  const int bRow = lane & 15;                     // column within 16-wide tile
  const int bKsel = (lane < 16) ? 0 : 16;         // B: same k order as A (16h..16h+15)

  v8f acc0 = {}, acc1 = {}, acc2 = {}, acc3 = {};

#pragma unroll
  for (int k = 0; k < 256; k += 32) {
    v16h afrag = *(const v16h*)(Ap + k + koff);    // K koff..koff+15
    const _Float16* bbase = tileB + k + bKsel;
    v16h b0 = *(const v16h*)(bbase + (size_t)(bRow + 0)  * 256);
    v16h b1 = *(const v16h*)(bbase + (size_t)(bRow + 16) * 256);
    v16h b2 = *(const v16h*)(bbase + (size_t)(bRow + 32) * 256);
    v16h b3 = *(const v16h*)(bbase + (size_t)(bRow + 48) * 256);
    acc0 = __builtin_amdgcn_wmma_f32_16x16x32_f16(false, afrag, false, b0,
                                                  (short)0, acc0, false, false);
    acc1 = __builtin_amdgcn_wmma_f32_16x16x32_f16(false, afrag, false, b1,
                                                  (short)0, acc1, false, false);
    acc2 = __builtin_amdgcn_wmma_f32_16x16x32_f16(false, afrag, false, b2,
                                                  (short)0, acc2, false, false);
    acc3 = __builtin_amdgcn_wmma_f32_16x16x32_f16(false, afrag, false, b3,
                                                  (short)0, acc3, false, false);
  }

  // C/D layout: VGPR j -> M = m0 + j (+8 for lanes 16-31), N = n0+16t+lane%16
  const int rbase = m0 + ((lane < 16) ? 0 : 8);
  const int ccol = lane & 15;
  v8f accs[4] = {acc0, acc1, acc2, acc3};
#pragma unroll
  for (int t = 0; t < 4; ++t) {
    int col = n0 + 16 * t + ccol;
    float bv = bias ? bias[col] : 0.f;
#pragma unroll
    for (int j = 0; j < 8; ++j) {
      int row = rbase + j;
      if (row < M) C[(size_t)row * 256 + col] = accs[t][j] + bv;
    }
  }
}

// ---------------------------------------------------------------------------
// Edge kernels (grid.y = relation)
// ---------------------------------------------------------------------------
__global__ void edge_score_kernel(const int* __restrict__ src, const int* __restrict__ dst,
                                  const float* __restrict__ Kr, const float* __restrict__ Q,
                                  const float* __restrict__ rel_pri,
                                  float* __restrict__ score, float* __restrict__ maxv,
                                  int E, int N) {
  int r = blockIdx.y;
  int idx = blockIdx.x * blockDim.x + threadIdx.x;
  if (idx >= E * H_HEADS) return;
  int e = idx >> 3, hh = idx & 7;
  int dt = r & 1;
  int s = src[(size_t)r * E + e];
  int d = dst[(size_t)r * E + e];
  const float4* qp = (const float4*)(Q + ((size_t)dt * N + d) * D_DIM + hh * DK_DIM);
  const float4* kp = (const float4*)(Kr + ((size_t)r * N + s) * D_DIM + hh * DK_DIM);
  float acc = 0.f;
#pragma unroll
  for (int i = 0; i < 8; ++i) {
    float4 a = qp[i], b = kp[i];
    acc += a.x * b.x + a.y * b.y + a.z * b.z + a.w * b.w;
  }
  acc *= rel_pri[r * H_HEADS + hh] * 0.17677669529663687f;  // 1/sqrt(32)
  score[((size_t)r * E + e) * H_HEADS + hh] = acc;
  float* mp = maxv + ((size_t)dt * N + d) * H_HEADS + hh;
  if (acc >= 0.f) atomicMax((int*)mp, __float_as_int(acc));
  else            atomicMin((unsigned int*)mp, __float_as_uint(acc));
}

__global__ void edge_exp_kernel(const int* __restrict__ dst, float* __restrict__ score,
                                const float* __restrict__ maxv, float* __restrict__ den,
                                int E, int N) {
  int r = blockIdx.y;
  int idx = blockIdx.x * blockDim.x + threadIdx.x;
  if (idx >= E * H_HEADS) return;
  int e = idx >> 3, hh = idx & 7;
  int dt = r & 1;
  int d = dst[(size_t)r * E + e];
  size_t si = ((size_t)r * E + e) * H_HEADS + hh;
  float m = maxv[((size_t)dt * N + d) * H_HEADS + hh];
  float v = expf(score[si] - m);
  score[si] = v;
  atomicAdd(den + ((size_t)dt * N + d) * H_HEADS + hh, v);
}

__global__ void edge_agg_kernel(const int* __restrict__ src, const int* __restrict__ dst,
                                const float* __restrict__ score, const float* __restrict__ den,
                                const float* __restrict__ Vr, float* __restrict__ agg,
                                int E, int N) {
  int r = blockIdx.y;
  int idx = blockIdx.x * blockDim.x + threadIdx.x;
  if (idx >= E * H_HEADS) return;
  int e = idx >> 3, hh = idx & 7;
  int dt = r & 1;
  int s = src[(size_t)r * E + e];
  int d = dst[(size_t)r * E + e];
  float attn = score[((size_t)r * E + e) * H_HEADS + hh] /
               den[((size_t)dt * N + d) * H_HEADS + hh];
  const float4* vp = (const float4*)(Vr + ((size_t)r * N + s) * D_DIM + hh * DK_DIM);
  float* ap = agg + ((size_t)dt * N + d) * D_DIM + hh * DK_DIM;
#pragma unroll
  for (int i = 0; i < 8; ++i) {
    float4 v = vp[i];
    atomicAdd(ap + 4 * i + 0, v.x * attn);
    atomicAdd(ap + 4 * i + 1, v.y * attn);
    atomicAdd(ap + 4 * i + 2, v.z * attn);
    atomicAdd(ap + 4 * i + 3, v.w * attn);
  }
}

// ---------------------------------------------------------------------------
// Output path
// ---------------------------------------------------------------------------
__global__ void blend_kernel(const float* __restrict__ trans, const float* __restrict__ h0,
                             const float* __restrict__ h1, const float* __restrict__ skip,
                             float* __restrict__ newh, int nPer) {
  int i = blockIdx.x * blockDim.x + threadIdx.x;
  if (i >= 2 * nPer) return;
  int t = (i < nPer) ? 0 : 1;
  float alpha = 1.f / (1.f + expf(-skip[t]));
  float hv = (t == 0) ? h0[i] : h1[i - nPer];
  newh[i] = trans[i] * alpha + hv * (1.f - alpha);
}

// Wc (256x9 f32 = 9216B) is staged into LDS with async global->LDS copies
// (ASYNCcnt path); every thread then re-reads it 256 times from LDS.
__global__ __launch_bounds__(256) void classify_kernel(
    const float* __restrict__ newh, const float* __restrict__ Wc,
    const float* __restrict__ bc, float* __restrict__ out, int Ntot) {
  __shared__ __align__(16) float WcLds[D_DIM * 9];
  const int tid = threadIdx.x;
  {
    unsigned ldsBase = (unsigned)(size_t)(void*)WcLds;
    const int chunks = (D_DIM * 9 * 4) / 16;  // 576 x 16B
    for (int c = tid; c < chunks; c += 256) {
      unsigned long long gaddr =
          (unsigned long long)(const void*)((const char*)Wc + (size_t)c * 16);
      unsigned laddr = ldsBase + (unsigned)c * 16u;
      asm volatile("global_load_async_to_lds_b128 %0, %1, off"
                   :: "v"(laddr), "v"(gaddr) : "memory");
    }
    asm volatile("s_wait_asynccnt 0" ::: "memory");
  }
  __syncthreads();

  int idx = blockIdx.x * blockDim.x + threadIdx.x;
  if (idx >= Ntot * 9) return;
  int c = idx % 9;
  int n = idx / 9;
  const float* hp = newh + (size_t)n * D_DIM;
  float acc = bc[c];
  for (int kk = 0; kk < D_DIM; ++kk) acc += hp[kk] * WcLds[kk * 9 + c];
  out[idx] = acc;
}

// ---------------------------------------------------------------------------
extern "C" void kernel_launch(void* const* d_in, const int* in_sizes, int n_in,
                              void* d_out, int out_size, void* d_ws, size_t ws_size,
                              hipStream_t stream) {
  (void)n_in; (void)out_size; (void)ws_size;
  const float* h0      = (const float*)d_in[0];
  const float* h1      = (const float*)d_in[1];
  const int*   src     = (const int*)d_in[2];
  const int*   dst     = (const int*)d_in[3];
  const float* Wk      = (const float*)d_in[4];
  const float* Wq      = (const float*)d_in[5];
  const float* Wv      = (const float*)d_in[6];
  const float* Wa      = (const float*)d_in[7];
  const float* bk      = (const float*)d_in[8];
  const float* bq      = (const float*)d_in[9];
  const float* bv      = (const float*)d_in[10];
  const float* ba      = (const float*)d_in[11];
  const float* rel_pri = (const float*)d_in[12];
  const float* rel_att = (const float*)d_in[13];
  const float* rel_msg = (const float*)d_in[14];
  const float* skip    = (const float*)d_in[15];
  const float* Wc      = (const float*)d_in[16];
  const float* bc      = (const float*)d_in[17];
  float* out = (float*)d_out;

  const int D = D_DIM, Hh = H_HEADS, R = 4;
  const int N = in_sizes[0] / D;    // 30000 per type
  const int E = in_sizes[2] / R;    // 240000
  const int Ntot = 2 * N;

  // workspace carve-out
  char* ws = (char*)d_ws;
  size_t off = 0;
  auto alloc = [&](size_t bytes) -> char* {
    off = (off + 255) & ~(size_t)255;
    char* p = ws + off;
    off += bytes;
    return p;
  };
  _Float16* h16  = (_Float16*)alloc((size_t)Ntot * D * 2);
  _Float16* WkrT = (_Float16*)alloc((size_t)R * D * D * 2);
  _Float16* WvrT = (_Float16*)alloc((size_t)R * D * D * 2);
  _Float16* WqT  = (_Float16*)alloc((size_t)2 * D * D * 2);
  _Float16* WaT  = (_Float16*)alloc((size_t)2 * D * D * 2);
  float* bkr     = (float*)alloc((size_t)R * D * 4);
  float* bvr     = (float*)alloc((size_t)R * D * 4);
  float* Kr      = (float*)alloc((size_t)R * N * D * 4);
  float* Vr      = (float*)alloc((size_t)R * N * D * 4);
  float* Qt      = (float*)alloc((size_t)2 * N * D * 4);
  float* score   = (float*)alloc((size_t)R * E * Hh * 4);
  float* maxv    = (float*)alloc((size_t)Ntot * Hh * 4);
  float* den     = (float*)alloc((size_t)Ntot * Hh * 4);
  float* agg     = (float*)alloc((size_t)Ntot * D * 4);
  _Float16* tf16 = (_Float16*)alloc((size_t)Ntot * D * 2);
  float* trans   = (float*)alloc((size_t)Ntot * D * 4);
  float* newh    = (float*)alloc((size_t)Ntot * D * 4);

  const int thr = 256;
  auto blocks = [&](long long n) { return (unsigned)((n + thr - 1) / thr); };

  // weight prep
  compose_rel_weight<<<blocks((long long)R * D * D), thr, 0, stream>>>(Wk, rel_att, WkrT, R * D * D);
  compose_rel_weight<<<blocks((long long)R * D * D), thr, 0, stream>>>(Wv, rel_msg, WvrT, R * D * D);
  compose_rel_bias<<<blocks(R * D), thr, 0, stream>>>(bk, rel_att, bkr, R * D);
  compose_rel_bias<<<blocks(R * D), thr, 0, stream>>>(bv, rel_msg, bvr, R * D);
  transpose_w_f16<<<blocks(2 * D * D), thr, 0, stream>>>(Wq, WqT, 2 * D * D);
  transpose_w_f16<<<blocks(2 * D * D), thr, 0, stream>>>(Wa, WaT, 2 * D * D);
  cvt2_f16<<<blocks((long long)Ntot * D), thr, 0, stream>>>(h0, h1, h16, N * D);

  // init reductions
  fill_f32<<<blocks((long long)Ntot * D), thr, 0, stream>>>(agg, 0.f, Ntot * D);
  fill_f32<<<blocks(Ntot * Hh), thr, 0, stream>>>(den, 0.f, Ntot * Hh);
  fill_f32<<<blocks(Ntot * Hh), thr, 0, stream>>>(maxv, -INFINITY, Ntot * Hh);

  // projection GEMMs (WMMA + TDM)
  const int strips = (N + 15) / 16;
  dim3 ggrid((strips + 3) / 4, D / 64);
  dim3 gblk(128);
  for (int r = 0; r < R; ++r) {
    int st = r >> 1;
    wmma_gemm<<<ggrid, gblk, 0, stream>>>(h16 + (size_t)st * N * D, WkrT + (size_t)r * D * D,
                                          bkr + r * D, Kr + (size_t)r * N * D, N);
    wmma_gemm<<<ggrid, gblk, 0, stream>>>(h16 + (size_t)st * N * D, WvrT + (size_t)r * D * D,
                                          bvr + r * D, Vr + (size_t)r * N * D, N);
  }
  for (int t = 0; t < 2; ++t)
    wmma_gemm<<<ggrid, gblk, 0, stream>>>(h16 + (size_t)t * N * D, WqT + (size_t)t * D * D,
                                          bq + t * D, Qt + (size_t)t * N * D, N);

  // edge attention
  dim3 egrid(blocks((long long)E * Hh), R);
  edge_score_kernel<<<egrid, thr, 0, stream>>>(src, dst, Kr, Qt, rel_pri, score, maxv, E, N);
  edge_exp_kernel<<<egrid, thr, 0, stream>>>(dst, score, maxv, den, E, N);
  edge_agg_kernel<<<egrid, thr, 0, stream>>>(src, dst, score, den, Vr, agg, E, N);

  // cross-relation mean (cnt=2) + a_linear GEMM + skip blend + classifier
  scale_cvt_f16<<<blocks((long long)Ntot * D), thr, 0, stream>>>(agg, tf16, 0.5f, Ntot * D);
  for (int t = 0; t < 2; ++t)
    wmma_gemm<<<ggrid, gblk, 0, stream>>>(tf16 + (size_t)t * N * D, WaT + (size_t)t * D * D,
                                          ba + t * D, trans + (size_t)t * N * D, N);
  blend_kernel<<<blocks((long long)Ntot * D), thr, 0, stream>>>(trans, h0, h1, skip, newh, N * D);
  classify_kernel<<<blocks((long long)Ntot * 9), thr, 0, stream>>>(newh, Wc, bc, out, Ntot);
}
